// BiLSTM_CRF_214748365219
// MI455X (gfx1250) — matrix operand order repaired and checked by reading, not compiled, not run on hardware
//
#include <hip/hip_runtime.h>
#include <hip/hip_bf16.h>
#include <stdint.h>

// ---------------------------------------------------------------------------
// BiLSTM-CRF for MI455X (gfx1250, wave32, WMMA)
//   k_gather   : embedding gather -> bf16, K padded 300->320
//   k_swz_wih  : W_ih (fwd|bwd, 2048 gates) -> WMMA B-fragment-ordered bf16
//   k_bias     : b_ih + b_hh combined, both dirs
//   k_swz_wtag : W_tag (12->16 padded) -> WMMA B-fragment-ordered bf16
//   k_whht     : W_hh^T interleaved [k/4][gate][4] f32 (float4 GEMV loads)
//   k_gemm_in  : [4096x320]x[320x2048] WMMA bf16 GEMM -> gate pre-activations
//   k_lstm     : sequential recurrence, 2 blocks (fwd/bwd), 1024 thr each,
//                async-to-LDS double buffering of pre-activation rows
//   k_feats    : [4096x512]x[512x16] WMMA bf16 GEMM -> CRF emission scores
//   k_viterbi  : 12-tag Viterbi + backtrace, single wave
// ---------------------------------------------------------------------------

typedef __attribute__((ext_vector_type(16))) __bf16 v16bf;
typedef __attribute__((ext_vector_type(8)))  __bf16 v8bf;
typedef __attribute__((ext_vector_type(8)))  float  v8f;

#define T_LEN 4096
#define EMB   300
#define EMBP  320            // padded K for WMMA (10 chunks of 32)
#define HD    256
#define G4    1024           // 4*HD gates per direction
#define NCOMB 2048           // both directions' gates
#define HIDC  512            // concat hidden
#define TAGS  12
#define TAGSP 16
#define START_TAG 10
#define STOP_TAG  11
#define NEGV  -10000.0f

#if defined(__has_builtin)
#if __has_builtin(__builtin_amdgcn_global_load_async_to_lds_b128) && \
    __has_builtin(__builtin_amdgcn_s_wait_asynccnt)
#define HAVE_ASYNC_LDS 1
#endif
#endif

#if defined(HAVE_ASYNC_LDS)
typedef int v4i_vs __attribute__((vector_size(16)));            // builtin's vec type
typedef __attribute__((address_space(3))) v4i_vs* lds_v4i_ptr;  // __shared__ int4*
#endif

__device__ __forceinline__ __bf16 f2bf(float f) {
  uint32_t u = __builtin_bit_cast(uint32_t, f);
  u = (u + 0x7FFFu + ((u >> 16) & 1u)) >> 16;   // round-to-nearest-even
  return __builtin_bit_cast(__bf16, (uint16_t)u);
}
__device__ __forceinline__ float sigf(float x) { return 1.0f / (1.0f + __expf(-x)); }

// ---- prep kernels ---------------------------------------------------------

__global__ void k_gather(const int* __restrict__ sent, const float* __restrict__ embed,
                         __bf16* __restrict__ xbf) {
  int t = blockIdx.x;                 // 0..4095
  int k = threadIdx.x;                // blockDim == EMBP (320)
  int row = sent[t];
  float v = (k < EMB) ? embed[(size_t)row * EMB + k] : 0.0f;
  xbf[(size_t)t * EMBP + k] = f2bf(v);
}

// layout: idx = ((ntile*10 + kchunk)*32 + lane)*16 + elem
//   lane<16: n = ntile*16 + lane,     K = k0 + elem        (elem 0..15)
//   lane>=16: n = ntile*16 + lane-16, K = k0 + 16 + elem
__global__ void k_swz_wih(const float* __restrict__ Wf, const float* __restrict__ Wb,
                          __bf16* __restrict__ dst) {
  int idx = blockIdx.x * blockDim.x + threadIdx.x;
  const int total = 128 * 10 * 32 * 16;
  if (idx >= total) return;
  int i    = idx & 15;
  int lane = (idx >> 4) & 31;
  int kc   = (idx >> 9) % 10;
  int nt   = idx / (16 * 32 * 10);
  int n = nt * 16 + (lane & 15);
  int k = kc * 32 + ((lane < 16) ? 0 : 16) + i;
  float v = 0.0f;
  if (k < EMB) v = (n < G4) ? Wf[(size_t)n * EMB + k] : Wb[(size_t)(n - G4) * EMB + k];
  dst[idx] = f2bf(v);
}

__global__ void k_bias(const float* __restrict__ bf1, const float* __restrict__ bf2,
                       const float* __restrict__ bb1, const float* __restrict__ bb2,
                       float* __restrict__ bias) {
  int n = blockIdx.x * blockDim.x + threadIdx.x;
  if (n >= NCOMB) return;
  bias[n] = (n < G4) ? (bf1[n] + bf2[n]) : (bb1[n - G4] + bb2[n - G4]);
}

__global__ void k_swz_wtag(const float* __restrict__ Wtag, __bf16* __restrict__ dst) {
  int idx = blockIdx.x * blockDim.x + threadIdx.x;
  const int total = 16 * 32 * 16;     // 16 k-chunks of 32 x 16 cols
  if (idx >= total) return;
  int i    = idx & 15;
  int lane = (idx >> 4) & 31;
  int kc   = idx >> 9;
  int n = lane & 15;
  int k = kc * 32 + ((lane < 16) ? 0 : 16) + i;
  float v = (n < TAGS) ? Wtag[(size_t)n * HIDC + k] : 0.0f;
  dst[idx] = f2bf(v);
}

// W_hh^T interleaved for float4 GEMV: dst[dir][k/4][gate][k%4]
__global__ void k_whht(const float* __restrict__ Wf, const float* __restrict__ Wb,
                       float* __restrict__ dst) {
  int idx = blockIdx.x * blockDim.x + threadIdx.x;
  if (idx >= 2 * HD * G4) return;
  int dir = idx / (HD * G4);
  int rem = idx % (HD * G4);
  int q  = rem / (G4 * 4);            // k/4
  int r2 = rem % (G4 * 4);
  int j  = r2 / 4;                    // gate
  int kl = r2 % 4;                    // k%4
  int k = q * 4 + kl;
  const float* W = dir ? Wb : Wf;
  dst[idx] = W[(size_t)j * HD + k];
}

// ---- WMMA input-projection GEMM: pre[t][n] = x[t][:] . Wih[n][:] + bias[n] --

__global__ void k_gemm_in(const __bf16* __restrict__ xbf, const __bf16* __restrict__ wih,
                          const float* __restrict__ bias, float* __restrict__ pre) {
  int lane = threadIdx.x;             // one wave per 16x16 tile
  int t0 = blockIdx.x * 16;
  int nt = blockIdx.y;                // 0..127 -> n0 = nt*16
  int m  = lane & 15;
  int kbaseA = (lane < 16) ? 0 : 16;  // contiguous K layout (k = 16h + s)
  const __bf16* arow  = xbf + (size_t)(t0 + m) * EMBP;
  const __bf16* bbase = wih + (size_t)nt * (10 * 32 * 16) + lane * 16;
  v8f acc = {};
  for (int kc = 0; kc < 10; ++kc) {
    int k0 = kc * 32;
    v8bf lo = *(const v8bf*)(arow + k0 + kbaseA);        // K = k0+kb .. +7
    v8bf hi = *(const v8bf*)(arow + k0 + kbaseA + 8);    // K = k0+kb+8 .. +15
    v16bf a;
#pragma unroll
    for (int i = 0; i < 8; ++i) { a[i] = lo[i]; a[8 + i] = hi[i]; }
    v16bf b = *(const v16bf*)(bbase + kc * (32 * 16));   // pre-swizzled fragment
    if (kc + 1 < 10) __builtin_prefetch(bbase + (kc + 1) * (32 * 16), 0, 0);
    acc = __builtin_amdgcn_wmma_f32_16x16x32_bf16(false, a, false, b,
                                                  (short)0, acc, false, false);
  }
  int cn = lane & 15;
  int mb = (lane < 16) ? 0 : 8;       // ISA f32 C layout
  int n0 = nt * 16;
  float bv = bias[n0 + cn];
#pragma unroll
  for (int r = 0; r < 8; ++r)
    pre[(size_t)(t0 + mb + r) * NCOMB + n0 + cn] = acc[r] + bv;
}

// ---- sequential LSTM recurrence (block 0 = forward, block 1 = backward) ----

__global__ void __launch_bounds__(1024)
k_lstm(const float* __restrict__ pre, const float* __restrict__ whhT,
       const float* __restrict__ h0, const float* __restrict__ c0,
       float* __restrict__ hf, float* __restrict__ hb) {
  int dir = blockIdx.x;
  int j = threadIdx.x;                // gate index 0..1023
  __shared__ float h_sh[HD];
  __shared__ float g_sh[G4];
#if defined(HAVE_ASYNC_LDS)
  __shared__ float pre_buf[2][G4];    // double-buffered gate pre-activations
#endif
  const float4* wt4 = (const float4*)(whhT + (size_t)dir * HD * G4);
  const float* prow = pre + (size_t)dir * G4;
  float* hout = dir ? hb : hf;
  float c = 0.0f;
  if (j < HD) { h_sh[j] = h0[dir * HD + j]; c = c0[dir * HD + j]; }
#if defined(HAVE_ASYNC_LDS)
  if (j < HD) {                       // prologue: stage row for step 0
    int t0 = dir ? (T_LEN - 1) : 0;
    __builtin_amdgcn_global_load_async_to_lds_b128(
        (v4i_vs*)(prow + (size_t)t0 * NCOMB + j * 4),
        (lds_v4i_ptr)(&pre_buf[0][j * 4]), 0, 0);
  }
  int cb = 0;
#endif
  __syncthreads();
  for (int s = 0; s < T_LEN; ++s) {
    int t = dir ? (T_LEN - 1 - s) : s;
#if defined(HAVE_ASYNC_LDS)
    if (j < HD && (s + 1) < T_LEN) {  // stage next row while GEMV runs
      int tn = dir ? (T_LEN - 2 - s) : (s + 1);
      __builtin_amdgcn_global_load_async_to_lds_b128(
          (v4i_vs*)(prow + (size_t)tn * NCOMB + j * 4),
          (lds_v4i_ptr)(&pre_buf[cb ^ 1][j * 4]), 0, 0);
    }
#endif
    float acc = 0.0f;
#pragma unroll 4
    for (int q = 0; q < HD / 4; ++q) {
      float4 w  = wt4[(size_t)q * G4 + j];
      float4 hv = *(const float4*)&h_sh[q * 4];
      acc += w.x * hv.x + w.y * hv.y + w.z * hv.z + w.w * hv.w;
    }
#if defined(HAVE_ASYNC_LDS)
    g_sh[j] = acc;
    if ((s + 1) < T_LEN) __builtin_amdgcn_s_wait_asynccnt(1);
    else                 __builtin_amdgcn_s_wait_asynccnt(0);
#else
    g_sh[j] = acc + prow[(size_t)t * NCOMB + j];
#endif
    __syncthreads();                  // publish g_sh (and pre_buf[cb])
    if (j < HD) {                     // PyTorch gate order i,f,g,o
#if defined(HAVE_ASYNC_LDS)
      float ig = sigf (g_sh[j]          + pre_buf[cb][j]);
      float fg = sigf (g_sh[HD + j]     + pre_buf[cb][HD + j]);
      float gg = tanhf(g_sh[2 * HD + j] + pre_buf[cb][2 * HD + j]);
      float og = sigf (g_sh[3 * HD + j] + pre_buf[cb][3 * HD + j]);
#else
      float ig = sigf (g_sh[j]);
      float fg = sigf (g_sh[HD + j]);
      float gg = tanhf(g_sh[2 * HD + j]);
      float og = sigf (g_sh[3 * HD + j]);
#endif
      c = fg * c + ig * gg;
      float h = og * tanhf(c);
      h_sh[j] = h;
      hout[(size_t)t * HD + j] = h;
    }
#if defined(HAVE_ASYNC_LDS)
    cb ^= 1;
#endif
    __syncthreads();
  }
}

// ---- WMMA emission GEMM: feats[t][tag] = concat(hf,hb)[t] . Wtag[tag] + b --

__global__ void k_feats(const float* __restrict__ hf, const float* __restrict__ hb,
                        const __bf16* __restrict__ wtag, const float* __restrict__ btag,
                        float* __restrict__ feats) {
  int lane = threadIdx.x;
  int t0 = blockIdx.x * 16;
  int m  = lane & 15;
  int kbaseA = (lane < 16) ? 0 : 16;
  v8f acc = {};
  for (int kc = 0; kc < 16; ++kc) {
    int k0 = kc * 32;
    v16bf a;
#pragma unroll
    for (int i = 0; i < 16; ++i) {
      int kk = k0 + kbaseA + i;
      float v = (kk < HD) ? hf[(size_t)(t0 + m) * HD + kk]
                          : hb[(size_t)(t0 + m) * HD + (kk - HD)];
      a[i] = f2bf(v);
    }
    v16bf b = *(const v16bf*)(wtag + (size_t)(kc * 32 + lane) * 16);
    acc = __builtin_amdgcn_wmma_f32_16x16x32_bf16(false, a, false, b,
                                                  (short)0, acc, false, false);
  }
  int cn = lane & 15;
  int mb = (lane < 16) ? 0 : 8;
  float bv = (cn < TAGS) ? btag[cn] : 0.0f;
#pragma unroll
  for (int r = 0; r < 8; ++r)
    feats[(size_t)(t0 + mb + r) * TAGSP + cn] = acc[r] + bv;
}

// ---- Viterbi decode, single wave ------------------------------------------

__global__ void k_viterbi(const float* __restrict__ feats, const float* __restrict__ trans,
                          int* __restrict__ bps, float* __restrict__ out) {
  int lane = threadIdx.x;             // 32 lanes, 12 active
  __shared__ float tr[TAGS * TAGS];
  __shared__ float fv[TAGS];
  for (int i = lane; i < TAGS * TAGS; i += 32) tr[i] = trans[i];
  if (lane < TAGS) fv[lane] = (lane == START_TAG) ? 0.0f : NEGV;
  __syncthreads();
  for (int t = 0; t < T_LEN; ++t) {
    float best = -3.0e38f; int bp = 0;
    if (lane < TAGS) {
#pragma unroll
      for (int f = 0; f < TAGS; ++f) {           // first max wins (argmax semantics)
        float v = fv[f] + tr[lane * TAGS + f];
        if (v > best) { best = v; bp = f; }
      }
    }
    __syncthreads();
    if (lane < TAGS) {
      fv[lane] = best + feats[(size_t)t * TAGSP + lane];
      bps[t * TAGS + lane] = bp;
    }
    __syncthreads();
  }
  if (lane == 0) {
    float bestv = -3.0e38f; int bi = 0;
    for (int i = 0; i < TAGS; ++i) {
      float v = fv[i] + tr[STOP_TAG * TAGS + i];
      if (v > bestv) { bestv = v; bi = i; }
    }
    out[0] = bestv;
    int tag = bi;
    for (int t = T_LEN - 1; t >= 0; --t) {
      out[1 + t] = (float)tag;
      tag = bps[t * TAGS + tag];
    }
  }
}

// ---------------------------------------------------------------------------

extern "C" void kernel_launch(void* const* d_in, const int* in_sizes, int n_in,
                              void* d_out, int out_size, void* d_ws, size_t ws_size,
                              hipStream_t stream) {
  (void)in_sizes; (void)n_in; (void)out_size; (void)ws_size;
  const int*   sent  = (const int*)d_in[0];
  const float* h0    = (const float*)d_in[1];
  const float* c0    = (const float*)d_in[2];
  const float* embed = (const float*)d_in[3];
  const float* Wih_f = (const float*)d_in[4];
  const float* Whh_f = (const float*)d_in[5];
  const float* bih_f = (const float*)d_in[6];
  const float* bhh_f = (const float*)d_in[7];
  const float* Wih_b = (const float*)d_in[8];
  const float* Whh_b = (const float*)d_in[9];
  const float* bih_b = (const float*)d_in[10];
  const float* bhh_b = (const float*)d_in[11];
  const float* Wtag  = (const float*)d_in[12];
  const float* btag  = (const float*)d_in[13];
  const float* trans = (const float*)d_in[14];

  char* ws = (char*)d_ws;
  size_t off = 0;
  auto alloc = [&](size_t bytes) -> void* {
    void* p = ws + off;
    off += (bytes + 255) & ~(size_t)255;
    return p;
  };
  __bf16* xbf   = (__bf16*)alloc((size_t)T_LEN * EMBP * 2);       // 2.6 MB
  __bf16* wih   = (__bf16*)alloc((size_t)128 * 10 * 32 * 16 * 2); // 1.3 MB
  float*  bias  = (float*) alloc((size_t)NCOMB * 4);
  __bf16* wtag  = (__bf16*)alloc((size_t)16 * 32 * 16 * 2);
  float*  whhT  = (float*) alloc((size_t)2 * HD * G4 * 4);        // 2 MB
  float*  pre   = (float*) alloc((size_t)T_LEN * NCOMB * 4);      // 33.5 MB
  float*  hfb   = (float*) alloc((size_t)T_LEN * HD * 4);         // 4.2 MB
  float*  hbb   = (float*) alloc((size_t)T_LEN * HD * 4);         // 4.2 MB
  float*  feats = (float*) alloc((size_t)T_LEN * TAGSP * 4);
  int*    bps   = (int*)   alloc((size_t)T_LEN * TAGS * 4);

  k_gather<<<T_LEN, EMBP, 0, stream>>>(sent, embed, xbf);
  { int tot = 128 * 10 * 32 * 16;
    k_swz_wih<<<(tot + 255) / 256, 256, 0, stream>>>(Wih_f, Wih_b, wih); }
  k_bias<<<(NCOMB + 255) / 256, 256, 0, stream>>>(bih_f, bhh_f, bih_b, bhh_b, bias);
  { int tot = 16 * 32 * 16;
    k_swz_wtag<<<(tot + 255) / 256, 256, 0, stream>>>(Wtag, wtag); }
  { int tot = 2 * HD * G4;
    k_whht<<<(tot + 255) / 256, 256, 0, stream>>>(Whh_f, Whh_b, whhT); }

  { dim3 g(T_LEN / 16, NCOMB / 16);                 // 256 x 128 tiles, 1 wave each
    k_gemm_in<<<g, 32, 0, stream>>>(xbf, wih, bias, pre); }

  k_lstm<<<2, G4, 0, stream>>>(pre, whhT, h0, c0, hfb, hbb);

  k_feats<<<T_LEN / 16, 32, 0, stream>>>(hfb, hbb, wtag, btag, feats);

  k_viterbi<<<1, 32, 0, stream>>>(feats, trans, bps, (float*)d_out);
}
